// MambaLayer_7069516169525
// MI455X (gfx1250) — matrix operand order repaired and checked by reading, not compiled, not run on hardware
//
#include <hip/hip_runtime.h>
#include <hip/hip_bf16.h>

// ---------------------------------------------------------------------------
// Mamba block for MI455X (gfx1250): bf16 WMMA GEMMs with async-LDS staged B
// tiles (double buffered), register-pipelined A fragments, and a
// register-resident selective scan.
// ---------------------------------------------------------------------------

#define D_MODEL 768
#define D_STATE 16
#define D_CONV  4
#define D_INNER 1536
#define DT_RANK 48
#define DT_PAD  64
#define XDBL_W  80          // DT_RANK + 2*D_STATE
#define HID     3072
#define B_SZ    4
#define SEQ     2048
#define NTOK    (B_SZ * SEQ)   // 8192

#if defined(__has_builtin)
#  if __has_builtin(__builtin_amdgcn_global_load_async_to_lds_b128) && \
      __has_builtin(__builtin_amdgcn_s_wait_asynccnt)
#    define USE_ASYNC_LDS 1
#  endif
#endif
#ifndef USE_ASYNC_LDS
#  define USE_ASYNC_LDS 0
#endif

typedef __bf16 bf16_t;
typedef __attribute__((ext_vector_type(16))) __bf16 v16bf;
typedef __attribute__((ext_vector_type(8)))  float  v8f;
typedef __attribute__((ext_vector_type(4)))  int    v4i;

typedef __attribute__((address_space(1))) v4i* as1_v4i_ptr;
typedef __attribute__((address_space(3))) v4i* as3_v4i_ptr;

union FragU {
    v16bf v;
    float4 q[2];
};

__device__ __forceinline__ unsigned short f2bf(float f) {
    union { float f; unsigned u; } v; v.f = f;
    unsigned r = v.u + 0x7FFFu + ((v.u >> 16) & 1u);   // round-to-nearest-even
    return (unsigned short)(r >> 16);
}

// A-matrix 16x32 bf16 fragment, contiguous k order (same as B, not the ISA table order): lane holds row M = lane%16.
// lanes 0-15: K = [k0..k0+15]; lanes 16-31: K = [k0+16..k0+31].
__device__ __forceinline__ v16bf load_frag_a(const unsigned short* __restrict__ p,
                                             int ld, int row, int k0, int lane) {
    const unsigned short* base = p + (size_t)row * ld + k0 + ((lane >> 4) & 1) * 16;
    FragU f;
    f.q[0] = *(const float4*)(base);
    f.q[1] = *(const float4*)(base + 8);
    return f.v;
}

// B-matrix 32x16 bf16 fragment from global: lane holds column N = lane%16
// (a row of W). lanes 0-15: K=[k0..k0+15]; lanes 16-31: K=[k0+16..k0+31].
__device__ __forceinline__ v16bf load_frag_b(const unsigned short* __restrict__ p,
                                             int ld, int row, int k0, int lane) {
    const unsigned short* base = p + (size_t)row * ld + k0 + ((lane >> 4) & 1) * 16;
    FragU f;
    f.q[0] = *(const float4*)(base);
    f.q[1] = *(const float4*)(base + 8);
    return f.v;
}

// Same B fragment but from an LDS-staged [64][32] bf16 tile.
__device__ __forceinline__ v16bf load_frag_b_lds(const unsigned short* buf,
                                                 int trow, int lane) {
    const unsigned short* base = buf + trow * 32 + ((lane >> 4) & 1) * 16;
    FragU f;
    f.q[0] = *(const float4*)(base);
    f.q[1] = *(const float4*)(base + 8);
    return f.v;
}

__device__ __forceinline__ float softplus_f(float x) {
    return (x > 20.f) ? x : log1pf(__expf(x));
}

// ---------------------------------------------------------------------------
// Generic bf16 WMMA GEMM:  C[M,N] = A[M,K] * W[N,K]^T  (both row-major, K-major)
// Wave computes a 16x64 tile (4 accumulators, A fragment reused); 8 waves
// stack 128 rows. B block-tile (64x32) is staged in LDS via async-LDS loads,
// double buffered; A fragments are double buffered in registers.
// EPI: 0 = f32 store, 1 = softplus(acc+bias) f32, 2 = acc+resid f32,
//      3 = relu(acc+bias) bf16, 4 = acc+bias+resid f32.
// ---------------------------------------------------------------------------
template <int EPI>
__global__ __launch_bounds__(256) void gemm_bf16_wmma(
    const unsigned short* __restrict__ A, const unsigned short* __restrict__ Bw,
    int M, int N, int K,
    float* __restrict__ outF, unsigned short* __restrict__ outB,
    const float* __restrict__ bias, const float* __restrict__ resid)
{
    const int lane = threadIdx.x & 31;
    const int wave = threadIdx.x >> 5;
    const int m0 = blockIdx.y * 128 + wave * 16;
    const int n0 = blockIdx.x * 64;

#if USE_ASYNC_LDS
    __shared__ __align__(16) unsigned short bstage[2][64 * 32];
    auto stage = [&](int buf, int kk) {
        const int row = threadIdx.x >> 2;            // 0..63
        const int ch  = (threadIdx.x & 3) * 8;       // 8 bf16 = 16B chunk
        int grow = n0 + row;
        if (grow >= N) grow = N - 1;                 // clamp (stores guarded)
        const unsigned short* gsrc = Bw + (size_t)grow * K + kk + ch;
        unsigned short* ldst = &bstage[buf][row * 32 + ch];
        __builtin_amdgcn_global_load_async_to_lds_b128(
            (as1_v4i_ptr)(v4i*)(void*)const_cast<unsigned short*>(gsrc),
            (as3_v4i_ptr)(v4i*)(void*)ldst, 0, 0);
    };
#endif

    v8f acc[4];
    #pragma unroll
    for (int t = 0; t < 4; ++t)
        #pragma unroll
        for (int e = 0; e < 8; ++e) acc[t][e] = 0.f;

    int arow = m0 + (lane & 15);
    if (arow >= M) arow = M - 1;
    const int nsteps = K / 32;

#if USE_ASYNC_LDS
    stage(0, 0);
#endif
    v16bf a_cur = load_frag_a(A, K, arow, 0, lane);

    for (int i = 0; i < nsteps; ++i) {
        const int k0 = i * 32;
#if USE_ASYNC_LDS
        __builtin_amdgcn_s_wait_asynccnt(0);         // own stage for step i done
        __syncthreads();                             // whole tile visible
        if (i + 1 < nsteps) stage((i + 1) & 1, k0 + 32);
#endif
        v16bf a_nxt = a_cur;
        if (i + 1 < nsteps) a_nxt = load_frag_a(A, K, arow, k0 + 32, lane);

        #pragma unroll
        for (int t = 0; t < 4; ++t) {
#if USE_ASYNC_LDS
            const int trow = t * 16 + (lane & 15);
            v16bf b = load_frag_b_lds(&bstage[i & 1][0], trow, lane);
#else
            int brow = n0 + t * 16 + (lane & 15);
            if (brow >= N) brow = N - 1;
            v16bf b = load_frag_b(Bw, K, brow, k0, lane);
#endif
            acc[t] = __builtin_amdgcn_wmma_f32_16x16x32_bf16(
                false, a_cur, false, b, (short)0, acc[t], false, false);
        }
#if USE_ASYNC_LDS
        __syncthreads();     // all waves done reading buf[i&1] before reuse
#endif
        a_cur = a_nxt;
    }

    // D layout: VGPR r, lane l -> row = m0 + r + 8*(l>=16), col = n0+16t+(l%16)
    const int coln  = lane & 15;
    const int rbase = m0 + ((lane >> 4) & 1) * 8;
    #pragma unroll
    for (int t = 0; t < 4; ++t) {
        const int col = n0 + t * 16 + coln;
        if (col >= N) continue;
        #pragma unroll
        for (int r = 0; r < 8; ++r) {
            const int row = rbase + r;
            if (row >= M) continue;
            const size_t idx = (size_t)row * N + col;
            const float v = acc[t][r];
            if (EPI == 0) {
                outF[idx] = v;
            } else if (EPI == 1) {
                outF[idx] = softplus_f(v + bias[col]);
            } else if (EPI == 2) {
                outF[idx] = v + resid[idx];
            } else if (EPI == 3) {
                float s = v + bias[col];
                outB[idx] = f2bf(s > 0.f ? s : 0.f);
            } else {
                outF[idx] = v + bias[col] + resid[idx];
            }
        }
    }
}

// ---------------------------------------------------------------------------
// LayerNorm over D=768 per token -> bf16 output. One 256-thread block / token.
// ---------------------------------------------------------------------------
__global__ __launch_bounds__(256) void ln_kernel(
    const float* __restrict__ x, const float* __restrict__ w,
    const float* __restrict__ b, unsigned short* __restrict__ out, int D)
{
    const int tok = blockIdx.x;
    const float* xr = x + (size_t)tok * D;
    __shared__ float red[2][8];
    float s = 0.f, s2 = 0.f;
    for (int i = threadIdx.x; i < D; i += 256) {
        float v = xr[i]; s += v; s2 += v * v;
    }
    #pragma unroll
    for (int off = 16; off > 0; off >>= 1) {
        s  += __shfl_down(s,  off, 32);
        s2 += __shfl_down(s2, off, 32);
    }
    const int lane = threadIdx.x & 31, wv = threadIdx.x >> 5;
    if (lane == 0) { red[0][wv] = s; red[1][wv] = s2; }
    __syncthreads();
    if (threadIdx.x == 0) {
        float a = 0.f, c = 0.f;
        #pragma unroll
        for (int i = 0; i < 8; ++i) { a += red[0][i]; c += red[1][i]; }
        red[0][0] = a; red[1][0] = c;
    }
    __syncthreads();
    const float mean = red[0][0] / D;
    const float var  = red[1][0] / D - mean * mean;
    const float inv  = rsqrtf(var + 1e-5f);
    for (int i = threadIdx.x; i < D; i += 256)
        out[(size_t)tok * D + i] = f2bf((xr[i] - mean) * inv * w[i] + b[i]);
}

// ---------------------------------------------------------------------------
// f32 -> bf16 convert with optional column zero-padding (weights + dt_r pack).
// ---------------------------------------------------------------------------
__global__ void cvt_bf16_kernel(const float* __restrict__ src,
                                unsigned short* __restrict__ dst,
                                int rows, int sc, int dc, int sstride)
{
    const long idx = (long)blockIdx.x * blockDim.x + threadIdx.x;
    if (idx >= (long)rows * dc) return;
    const int r = (int)(idx / dc), j = (int)(idx % dc);
    dst[idx] = (j < sc) ? f2bf(src[(size_t)r * sstride + j]) : (unsigned short)0;
}

// ---------------------------------------------------------------------------
// Depthwise causal conv (width 4) + bias + SiLU over the u half of xz.
// ---------------------------------------------------------------------------
__global__ void conv_silu_kernel(const float* __restrict__ xz,
                                 const float* __restrict__ cw,
                                 const float* __restrict__ cb,
                                 float* __restrict__ uf,
                                 unsigned short* __restrict__ ub)
{
    const long idx = (long)blockIdx.x * blockDim.x + threadIdx.x;
    if (idx >= (long)NTOK * D_INNER) return;
    const int c = (int)(idx % D_INNER);
    const int tok = (int)(idx / D_INNER);
    const int l = tok % SEQ;
    float acc = cb[c];
    #pragma unroll
    for (int k = 0; k < D_CONV; ++k) {
        const int ll = l - (D_CONV - 1) + k;
        if (ll >= 0)
            acc += cw[c * D_CONV + k] * xz[(size_t)(tok - (D_CONV - 1) + k) * (2 * D_INNER) + c];
    }
    const float s = acc / (1.f + __expf(-acc));   // SiLU
    uf[idx] = s;
    ub[idx] = f2bf(s);
}

// ---------------------------------------------------------------------------
// Selective scan: thread owns one (batch, channel) pair, 16 states in regs.
// B_t / C_t broadcast through LDS each step. Fuses D-skip and z-gate.
// grid = (D_INNER/512, B_SZ), block = 512.
// ---------------------------------------------------------------------------
__global__ __launch_bounds__(512) void scan_kernel(
    const float* __restrict__ xdbl, const float* __restrict__ dt,
    const float* __restrict__ uact, const float* __restrict__ xz,
    const float* __restrict__ A_log, const float* __restrict__ D_skip,
    unsigned short* __restrict__ ybf)
{
    const int c = blockIdx.x * 512 + threadIdx.x;   // 0..1535
    const int b = blockIdx.y;
    __shared__ float bc[32];                        // [0:16)=B_t, [16:32)=C_t

    float Ac[D_STATE];
    #pragma unroll
    for (int n = 0; n < D_STATE; ++n) Ac[n] = -__expf(A_log[c * D_STATE + n]);
    const float Dsk = D_skip[c];

    float h[D_STATE];
    #pragma unroll
    for (int n = 0; n < D_STATE; ++n) h[n] = 0.f;

    for (int l = 0; l < SEQ; ++l) {
        const int tok = b * SEQ + l;
        if (threadIdx.x < 32)
            bc[threadIdx.x] = xdbl[(size_t)tok * XDBL_W + DT_RANK + threadIdx.x];
        __syncthreads();

        const float dtt = dt[(size_t)tok * D_INNER + c];
        const float ut  = uact[(size_t)tok * D_INNER + c];
        const float du  = dtt * ut;
        float y = 0.f;
        #pragma unroll
        for (int n = 0; n < D_STATE; ++n) {
            const float dA = __expf(dtt * Ac[n]);
            h[n] = h[n] * dA + du * bc[n];
            y += h[n] * bc[16 + n];
        }
        const float z  = xz[(size_t)tok * (2 * D_INNER) + D_INNER + c];
        const float sz = z / (1.f + __expf(-z));
        ybf[(size_t)tok * D_INNER + c] = f2bf((y + ut * Dsk) * sz);
        __syncthreads();
    }
}

// ---------------------------------------------------------------------------
// Host-side orchestration.
// ---------------------------------------------------------------------------
static inline size_t align256(size_t x) { return (x + 255) & ~(size_t)255; }

extern "C" void kernel_launch(void* const* d_in, const int* in_sizes, int n_in,
                              void* d_out, int out_size, void* d_ws, size_t ws_size,
                              hipStream_t stream)
{
    const float* x      = (const float*)d_in[0];
    const float* ln1_w  = (const float*)d_in[1];
    const float* ln1_b  = (const float*)d_in[2];
    const float* W_in   = (const float*)d_in[3];
    const float* conv_w = (const float*)d_in[4];
    const float* conv_b = (const float*)d_in[5];
    const float* W_x    = (const float*)d_in[6];
    const float* W_dt   = (const float*)d_in[7];
    const float* b_dt   = (const float*)d_in[8];
    const float* A_log  = (const float*)d_in[9];
    const float* D_skip = (const float*)d_in[10];
    const float* W_out  = (const float*)d_in[11];
    const float* ln2_w  = (const float*)d_in[12];
    const float* ln2_b  = (const float*)d_in[13];
    const float* W1     = (const float*)d_in[14];
    const float* b1     = (const float*)d_in[15];
    const float* W2     = (const float*)d_in[16];
    const float* b2     = (const float*)d_in[17];
    float* out = (float*)d_out;

    char* ws = (char*)d_ws;
    size_t off = 0;
    auto alloc = [&](size_t bytes) { char* p = ws + off; off = align256(off + bytes); return p; };

    unsigned short* lnxb  = (unsigned short*)alloc((size_t)NTOK * D_MODEL * 2);
    unsigned short* Winb  = (unsigned short*)alloc((size_t)(2 * D_INNER) * D_MODEL * 2);
    float*          xz    = (float*)         alloc((size_t)NTOK * 2 * D_INNER * 4);
    float*          uactf = (float*)         alloc((size_t)NTOK * D_INNER * 4);
    unsigned short* uactb = (unsigned short*)alloc((size_t)NTOK * D_INNER * 2);
    unsigned short* Wxb   = (unsigned short*)alloc((size_t)XDBL_W * D_INNER * 2);
    float*          xdbl  = (float*)         alloc((size_t)NTOK * XDBL_W * 4);
    unsigned short* dtrb  = (unsigned short*)alloc((size_t)NTOK * DT_PAD * 2);
    unsigned short* Wdtb  = (unsigned short*)alloc((size_t)D_INNER * DT_PAD * 2);
    float*          dtf   = (float*)         alloc((size_t)NTOK * D_INNER * 4);
    unsigned short* ybf   = (unsigned short*)alloc((size_t)NTOK * D_INNER * 2);
    unsigned short* Woutb = (unsigned short*)alloc((size_t)D_MODEL * D_INNER * 2);
    float*          hbuf  = (float*)         alloc((size_t)NTOK * D_MODEL * 4);
    unsigned short* ln2b  = (unsigned short*)alloc((size_t)NTOK * D_MODEL * 2);
    unsigned short* W1b   = (unsigned short*)alloc((size_t)HID * D_MODEL * 2);
    unsigned short* W2b   = (unsigned short*)alloc((size_t)D_MODEL * HID * 2);
    unsigned short* act1b = (unsigned short*)alloc((size_t)NTOK * HID * 2);
    (void)ws_size; (void)n_in; (void)in_sizes; (void)out_size;

    auto cvt = [&](const float* src, unsigned short* dst, int rows, int sc, int dc, int ss) {
        long n = (long)rows * dc;
        cvt_bf16_kernel<<<dim3((unsigned)((n + 255) / 256)), 256, 0, stream>>>(src, dst, rows, sc, dc, ss);
    };

    // 1. Weights -> bf16 (W_dt zero-padded K 48->64).
    cvt(W_in,  Winb,  2 * D_INNER, D_MODEL, D_MODEL, D_MODEL);
    cvt(W_x,   Wxb,   XDBL_W,      D_INNER, D_INNER, D_INNER);
    cvt(W_dt,  Wdtb,  D_INNER,     DT_RANK, DT_PAD,  DT_RANK);
    cvt(W_out, Woutb, D_MODEL,     D_INNER, D_INNER, D_INNER);
    cvt(W1,    W1b,   HID,         D_MODEL, D_MODEL, D_MODEL);
    cvt(W2,    W2b,   D_MODEL,     HID,     HID,     HID);

    // 2. LN1 -> bf16.
    ln_kernel<<<NTOK, 256, 0, stream>>>(x, ln1_w, ln1_b, lnxb, D_MODEL);

    // 3. xz = ln(x) @ W_in^T  [8192 x 3072], K=768.
    gemm_bf16_wmma<0><<<dim3((2 * D_INNER) / 64, NTOK / 128), 256, 0, stream>>>(
        lnxb, Winb, NTOK, 2 * D_INNER, D_MODEL, xz, nullptr, nullptr, nullptr);

    // 4. Depthwise causal conv + SiLU on u half.
    {
        long n = (long)NTOK * D_INNER;
        conv_silu_kernel<<<dim3((unsigned)((n + 255) / 256)), 256, 0, stream>>>(
            xz, conv_w, conv_b, uactf, uactb);
    }

    // 5. x_dbl = u @ W_x^T  [8192 x 80], K=1536.
    gemm_bf16_wmma<0><<<dim3((XDBL_W + 63) / 64, NTOK / 128), 256, 0, stream>>>(
        uactb, Wxb, NTOK, XDBL_W, D_INNER, xdbl, nullptr, nullptr, nullptr);

    // 6. Pack dt_r (first 48 cols of x_dbl) into zero-padded bf16 [8192 x 64].
    cvt(xdbl, dtrb, NTOK, DT_RANK, DT_PAD, XDBL_W);

    // 7. dt = softplus(dt_r @ W_dt^T + b_dt)  [8192 x 1536], K=64.
    gemm_bf16_wmma<1><<<dim3(D_INNER / 64, NTOK / 128), 256, 0, stream>>>(
        dtrb, Wdtb, NTOK, D_INNER, DT_PAD, dtf, nullptr, b_dt, nullptr);

    // 8. Selective scan (fuses D-skip + z-gate) -> y bf16.
    scan_kernel<<<dim3(D_INNER / 512, B_SZ), 512, 0, stream>>>(
        xdbl, dtf, uactf, xz, A_log, D_skip, ybf);

    // 9. h = x + y @ W_out^T  [8192 x 768], K=1536.
    gemm_bf16_wmma<2><<<dim3(D_MODEL / 64, NTOK / 128), 256, 0, stream>>>(
        ybf, Woutb, NTOK, D_MODEL, D_INNER, hbuf, nullptr, nullptr, x);

    // 10. LN2 -> bf16.
    ln_kernel<<<NTOK, 256, 0, stream>>>(hbuf, ln2_w, ln2_b, ln2b, D_MODEL);

    // 11. act1 = relu(ln2 @ W1^T + b1) -> bf16  [8192 x 3072], K=768.
    gemm_bf16_wmma<3><<<dim3(HID / 64, NTOK / 128), 256, 0, stream>>>(
        ln2b, W1b, NTOK, HID, D_MODEL, nullptr, act1b, b1, nullptr);

    // 12. out = h + act1 @ W2^T + b2  [8192 x 768], K=3072.
    gemm_bf16_wmma<4><<<dim3(D_MODEL / 64, NTOK / 128), 256, 0, stream>>>(
        act1b, W2b, NTOK, D_MODEL, HID, out, nullptr, b2, hbuf);
}
